// TextEmbedding_80487687127219
// MI455X (gfx1250) — matrix operand order repaired and checked by reading, not compiled, not run on hardware
//
#include <hip/hip_runtime.h>
#include <hip/hip_bf16.h>
#include <stdint.h>

// Problem constants (from the reference)
#define B_   16
#define TT_  256
#define S_   2048
#define D_   512
#define I_   1024
#define L_   4
#define M_   (B_ * S_)   // 32768 GEMM rows

// WMMA tile config (wave32, v_wmma_f32_16x16x32_bf16)
#define BM 128
#define BN 64
#define BK 32
#define ATILE ((BM / 16) * 32 * 16)   // A tile elems in fragment layout
#define BTILE ((BN / 16) * 32 * 16)   // B tile elems in fragment layout

typedef __attribute__((ext_vector_type(16))) __bf16 v16bf;
typedef __attribute__((ext_vector_type(8)))  float  v8f;

// ---------------------------------------------------------------------------
// LDS tiles are stored in *fragment layout*: [tile16][lane(32)][16 bf16] with
// each (tile,lane) chunk = 32 contiguous bytes. Per CDNA5 ISA 7.12.2:
//  A frag (16x32 bf16): lane = m + 16*h; elems 0-7 <-> k = 8h..8h+7,
//                       elems 8-15 <-> k = 16+8h..23+8h   (16B-contiguous runs)
//  B frag (32x16 bf16): lane = n + 16*h; same k order as A (elems 0-7 <-> 8h.., 8-15 <-> 16+8h..)
// So a fragment load is two ds_load_b128, and (for K-contiguous bf16 global
// data) each 16B sub-chunk is a straight GLOBAL_LOAD_ASYNC_TO_LDS_B128 copy.
// ---------------------------------------------------------------------------

__device__ inline void async_b128(uint32_t lds_off, const void* gptr) {
  asm volatile("global_load_async_to_lds_b128 %0, %1, off"
               :: "v"(lds_off), "v"((uint64_t)(uintptr_t)gptr)
               : "memory");
}
__device__ inline void wait_async0() {
  asm volatile("s_wait_asynccnt 0x0" ::: "memory");
}
__device__ inline uint32_t lds_addr(const void* p) {
  return (uint32_t)(uintptr_t)p;  // generic LDS ptr truncates to LDS offset
}

__device__ inline v16bf frag_ld(const __bf16* base, int tile, int lane) {
  return *(const v16bf*)(base + (tile * 32 + lane) * 16);
}

// Stage one BMxBK A tile (bf16, K-contiguous) into fragment-layout LDS: 512
// 16B chunks via async copies (2 per thread).
__device__ inline void stage_a_async(const __bf16* A, __bf16* Asb, int row0,
                                     int K, int k0, int tid) {
#pragma unroll
  for (int it = 0; it < 2; ++it) {
    const int ch = tid + it * 256;
    const int r = ch >> 2;        // 0..127
    const int kg = ch & 3;        // 8-elem k-group
    const int m = r & 15, t = r >> 4;
    const int lane16 = m + ((kg & 1) << 4);  // half = kg&1
    async_b128(lds_addr(&Asb[(t * 32 + lane16) * 16 + ((kg >> 1) << 3)]),
               A + (size_t)(row0 + r) * K + k0 + (kg << 3));
  }
}

// Stage one BKxBN B tile from transposed (K-contiguous) bf16 weights: 256
// 16B chunks via async copies (1 per thread).
__device__ inline void stage_b_async(const __bf16* Wt, __bf16* Bsb, int col0,
                                     int K, int k0, int tid) {
  const int colt = tid >> 6;     // 0..3
  const int rem = tid & 63;
  const int lane16 = rem >> 1;   // n' + 16*half
  const int c2 = rem & 1;
  const int half = lane16 >> 4;
  const int n_g = col0 + colt * 16 + (lane16 & 15);
  async_b128(lds_addr(&Bsb[(colt * 32 + lane16) * 16 + (c2 << 3)]),
             Wt + (size_t)n_g * K + k0 + half * 8 + (c2 << 4));
}

// ---------------------------------------------------------------------------
// Kernel 1: embedding + RoPE-table add + pad mask -> x (B,S,D) fp32
// ---------------------------------------------------------------------------
__global__ __launch_bounds__(256) void embed_kernel(
    const int* __restrict__ text, const float* __restrict__ table,
    const float* __restrict__ freq, float* __restrict__ X) {
  const int bs = blockIdx.x;
  const int b = bs >> 11, s = bs & (S_ - 1);
  int tp = 0;
  if (s < TT_) tp = text[b * TT_ + s] + 1;
  const size_t base = (size_t)bs * D_;
  const int t0 = threadIdx.x, t1 = threadIdx.x + 256;
  if (tp == 0) {
    X[base + t0] = 0.f;
    X[base + t1] = 0.f;
  } else {
    const float* e = table + (size_t)tp * D_;
    const float* f = freq + (size_t)s * D_;  // pos = min(s,4095) == s
    X[base + t0] = e[t0] + f[t0];
    X[base + t1] = e[t1] + f[t1];
  }
}

// ---------------------------------------------------------------------------
// Kernel 2: transpose + fp32->bf16 weight conversion (once per launch).
// in: (L, K, N) fp32 row-major; out: (L, N, K) bf16 (K contiguous per column)
// ---------------------------------------------------------------------------
__global__ __launch_bounds__(256) void transpose_cvt_kernel(
    const float* __restrict__ in, __bf16* __restrict__ out, int K, int N) {
  const int n = blockIdx.x % N, l = blockIdx.x / N;
  const float* src = in + (size_t)l * K * N + n;
  __bf16* dst = out + (size_t)blockIdx.x * K;
  for (int k = threadIdx.x; k < K; k += 256) dst[k] = (__bf16)src[(size_t)k * N];
}

// ---------------------------------------------------------------------------
// Kernel 3: depthwise conv (k=7, pad 3) + LayerNorm over D -> Hn (M,D) bf16
// ---------------------------------------------------------------------------
__global__ __launch_bounds__(256) void conv_ln_kernel(
    const float* __restrict__ X, const float* __restrict__ dw_w,
    const float* __restrict__ dw_b, const float* __restrict__ lng,
    const float* __restrict__ lnb, __bf16* __restrict__ Hn) {
  __shared__ float rs[256], rq[256];
  const int bs = blockIdx.x;
  const int b = bs >> 11, s = bs & (S_ - 1);
  const int tid = threadIdx.x;
  float vals[2];
#pragma unroll
  for (int j = 0; j < 2; ++j) {
    const int d = tid + j * 256;
    float acc = dw_b[d];
    const float* wd = dw_w + d * 7;
#pragma unroll
    for (int k = 0; k < 7; ++k) {
      const int ss = s + k - 3;
      if (ss >= 0 && ss < S_)
        acc += X[((size_t)b * S_ + ss) * D_ + d] * wd[k];
    }
    vals[j] = acc;
  }
  rs[tid] = vals[0] + vals[1];
  rq[tid] = vals[0] * vals[0] + vals[1] * vals[1];
  __syncthreads();
  for (int off = 128; off > 0; off >>= 1) {
    if (tid < off) { rs[tid] += rs[tid + off]; rq[tid] += rq[tid + off]; }
    __syncthreads();
  }
  const float mu  = rs[0] * (1.f / D_);
  const float var = rq[0] * (1.f / D_) - mu * mu;
  const float inv = rsqrtf(var + 1e-6f);
  const size_t base = (size_t)bs * D_;
#pragma unroll
  for (int j = 0; j < 2; ++j) {
    const int d = tid + j * 256;
    Hn[base + d] = (__bf16)((vals[j] - mu) * inv * lng[d] + lnb[d]);
  }
}

// ---------------------------------------------------------------------------
// Kernel 4: GEMM1  H = gelu(Hn(M,512)bf16 @ W1t^T + b1); gx[b,i] += sum_s H^2
// Double-buffered pipeline: tile it+1's async copies are issued before tile
// it's WMMAs, so the async DMA runs under the matrix pipe. One barrier/step.
// ---------------------------------------------------------------------------
__global__ __launch_bounds__(256) void gemm1_kernel(
    const __bf16* __restrict__ A, const __bf16* __restrict__ Wt,
    const float* __restrict__ bias, float* __restrict__ H,
    float* __restrict__ gx) {
  __shared__ __align__(32) __bf16 Asb[2][ATILE];  // 2 x 8 KB
  __shared__ __align__(32) __bf16 Bsb[2][BTILE];  // 2 x 4 KB
  const int K = D_, N = I_;
  const int NT = K / BK;  // 16
  const int tid = threadIdx.x;
  const int lane = tid & 31, wave = tid >> 5;
  const int wm = wave >> 1, wn = wave & 1;  // 4x2 wave grid -> 32x32 per wave
  const int row0 = blockIdx.x * BM;
  const int col0 = blockIdx.y * BN;

  v8f acc[2][2] = {};

  stage_a_async(A, Asb[0], row0, K, 0, tid);
  stage_b_async(Wt, Bsb[0], col0, K, 0, tid);
  wait_async0();
  __syncthreads();

  for (int it = 0; it < NT; ++it) {
    const int p = it & 1;
    if (it + 1 < NT) {  // overlap next tile's DMA with this tile's WMMAs
      stage_a_async(A, Asb[p ^ 1], row0, K, (it + 1) * BK, tid);
      stage_b_async(Wt, Bsb[p ^ 1], col0, K, (it + 1) * BK, tid);
    }
    v16bf af0 = frag_ld(Asb[p], wm * 2,     lane);
    v16bf af1 = frag_ld(Asb[p], wm * 2 + 1, lane);
    v16bf bf0 = frag_ld(Bsb[p], wn * 2,     lane);
    v16bf bf1 = frag_ld(Bsb[p], wn * 2 + 1, lane);
    acc[0][0] = __builtin_amdgcn_wmma_f32_16x16x32_bf16(false, af0, false, bf0, (short)0, acc[0][0], false, false);
    acc[0][1] = __builtin_amdgcn_wmma_f32_16x16x32_bf16(false, af0, false, bf1, (short)0, acc[0][1], false, false);
    acc[1][0] = __builtin_amdgcn_wmma_f32_16x16x32_bf16(false, af1, false, bf0, (short)0, acc[1][0], false, false);
    acc[1][1] = __builtin_amdgcn_wmma_f32_16x16x32_bf16(false, af1, false, bf1, (short)0, acc[1][1], false, false);
    wait_async0();
    __syncthreads();
  }

  const int half = lane >> 4, nlo = lane & 15;
  const int b = row0 >> 11;  // 128-row tile never crosses a batch (S%128==0)
#pragma unroll
  for (int i = 0; i < 2; ++i) {
#pragma unroll
    for (int j = 0; j < 2; ++j) {
      const int n = col0 + wn * 32 + j * 16 + nlo;
      const float bn = bias[n];
      const int mbase = row0 + wm * 32 + i * 16 + 8 * half;
      float sq = 0.f;
#pragma unroll
      for (int r = 0; r < 8; ++r) {
        float v = acc[i][j][r] + bn;
        v = 0.5f * v * (1.0f + erff(v * 0.70710678118654752f));  // exact gelu
        H[(size_t)(mbase + r) * N + n] = v;
        sq += v * v;
      }
      atomicAdd(&gx[b * I_ + n], sq);
    }
  }
}

// ---------------------------------------------------------------------------
// Kernel 5: per-batch GRN scale: nx[i] = sqrt(gx[i]) / (mean_i sqrt(gx)+1e-6)
// ---------------------------------------------------------------------------
__global__ __launch_bounds__(256) void grn_nx_kernel(float* __restrict__ gx) {
  __shared__ float rs[256];
  const int b = blockIdx.x, tid = threadIdx.x;
  float g[4];
  float lsum = 0.f;
#pragma unroll
  for (int j = 0; j < 4; ++j) {
    const int i = tid + j * 256;
    g[j] = sqrtf(gx[b * I_ + i]);
    lsum += g[j];
  }
  rs[tid] = lsum;
  __syncthreads();
  for (int off = 128; off > 0; off >>= 1) {
    if (tid < off) rs[tid] += rs[tid + off];
    __syncthreads();
  }
  const float inv = 1.f / (rs[0] * (1.f / I_) + 1e-6f);
#pragma unroll
  for (int j = 0; j < 4; ++j) {
    const int i = tid + j * 256;
    gx[b * I_ + i] = g[j] * inv;
  }
}

// ---------------------------------------------------------------------------
// Kernel 6: GEMM2  x = mask * ((H*(1+g*nx)+gb)(M,1024) @ W2t^T + b2 + x)
// B tiles: double-buffered async copies. A tiles: register-pipelined — H
// float2 loads for tile it+1 issue before tile it's WMMAs; the fused GRN
// scale + v_cvt_pk_bf16 + ds_store_b32 into fragment layout happen after.
// ---------------------------------------------------------------------------
__global__ __launch_bounds__(256) void gemm2_kernel(
    const float* __restrict__ H, const __bf16* __restrict__ Wt,
    const float* __restrict__ bias, const float* __restrict__ grn_g,
    const float* __restrict__ grn_b, const float* __restrict__ nx,
    const int* __restrict__ text, float* __restrict__ X) {
  __shared__ __align__(32) __bf16 Asb[2][ATILE];
  __shared__ __align__(32) __bf16 Bsb[2][BTILE];
  const int K = I_, N = D_;
  const int NT = K / BK;  // 32
  const int tid = threadIdx.x;
  const int lane = tid & 31, wave = tid >> 5;
  const int wm = wave >> 1, wn = wave & 1;
  const int row0 = blockIdx.x * BM;
  const int col0 = blockIdx.y * BN;
  const int bb = row0 >> 11;

  v8f acc[2][2] = {};

  // This thread's k-pair (kp, kp+1) and its fragment-chunk slot are invariant.
  const int kp = (tid & 15) << 1;
  const int h2 = (kp >> 3) & 1;
  const int e2 = (kp & 7) + ((kp >> 4) << 3);  // even elem index in 32B chunk

  // --- prologue: tile 0 ---
  stage_b_async(Wt, Bsb[0], col0, K, 0, tid);
  {
    const int i0 = kp;
    const float sc0 = 1.0f + grn_g[i0] * nx[bb * I_ + i0];
    const float sc1 = 1.0f + grn_g[i0 + 1] * nx[bb * I_ + i0 + 1];
    const float gb0 = grn_b[i0], gb1 = grn_b[i0 + 1];
#pragma unroll
    for (int u = 0; u < 8; ++u) {
      const int r = (tid + u * 256) >> 4;
      const float2 hv = *(const float2*)&H[(size_t)(row0 + r) * K + i0];
      union { __bf16 b[2]; uint32_t u32; } pk;
      pk.b[0] = (__bf16)(hv.x * sc0 + gb0);
      pk.b[1] = (__bf16)(hv.y * sc1 + gb1);
      *(uint32_t*)&Asb[0][((r >> 4) * 32 + (r & 15) + 16 * h2) * 16 + e2] = pk.u32;
    }
  }
  wait_async0();
  __syncthreads();

  for (int it = 0; it < NT; ++it) {
    const int p = it & 1;
    float2 hn[8];
    float sc0 = 0.f, sc1 = 0.f, gb0 = 0.f, gb1 = 0.f;
    if (it + 1 < NT) {
      stage_b_async(Wt, Bsb[p ^ 1], col0, K, (it + 1) * BK, tid);
      const int i1 = (it + 1) * BK + kp;
      sc0 = 1.0f + grn_g[i1] * nx[bb * I_ + i1];
      sc1 = 1.0f + grn_g[i1 + 1] * nx[bb * I_ + i1 + 1];
      gb0 = grn_b[i1];
      gb1 = grn_b[i1 + 1];
#pragma unroll
      for (int u = 0; u < 8; ++u) {  // issue loads now; consume after WMMAs
        const int r = (tid + u * 256) >> 4;
        hn[u] = *(const float2*)&H[(size_t)(row0 + r) * K + i1];
      }
    }

    v16bf af0 = frag_ld(Asb[p], wm * 2,     lane);
    v16bf af1 = frag_ld(Asb[p], wm * 2 + 1, lane);
    v16bf bf0 = frag_ld(Bsb[p], wn * 2,     lane);
    v16bf bf1 = frag_ld(Bsb[p], wn * 2 + 1, lane);
    acc[0][0] = __builtin_amdgcn_wmma_f32_16x16x32_bf16(false, af0, false, bf0, (short)0, acc[0][0], false, false);
    acc[0][1] = __builtin_amdgcn_wmma_f32_16x16x32_bf16(false, af0, false, bf1, (short)0, acc[0][1], false, false);
    acc[1][0] = __builtin_amdgcn_wmma_f32_16x16x32_bf16(false, af1, false, bf0, (short)0, acc[1][0], false, false);
    acc[1][1] = __builtin_amdgcn_wmma_f32_16x16x32_bf16(false, af1, false, bf1, (short)0, acc[1][1], false, false);

    if (it + 1 < NT) {
#pragma unroll
      for (int u = 0; u < 8; ++u) {
        const int r = (tid + u * 256) >> 4;
        union { __bf16 b[2]; uint32_t u32; } pk;
        pk.b[0] = (__bf16)(hn[u].x * sc0 + gb0);
        pk.b[1] = (__bf16)(hn[u].y * sc1 + gb1);
        *(uint32_t*)&Asb[p ^ 1][((r >> 4) * 32 + (r & 15) + 16 * h2) * 16 + e2] = pk.u32;
      }
    }
    wait_async0();
    __syncthreads();
  }

  const int half = lane >> 4, nlo = lane & 15;
#pragma unroll
  for (int i = 0; i < 2; ++i) {
#pragma unroll
    for (int j = 0; j < 2; ++j) {
      const int n = col0 + wn * 32 + j * 16 + nlo;
      const float bn = bias[n];
      const int mbase = row0 + wm * 32 + i * 16 + 8 * half;
#pragma unroll
      for (int r = 0; r < 8; ++r) {
        const int m = mbase + r;
        const int b = m >> 11, s = m & (S_ - 1);
        const bool keep = (s < TT_) && (text[b * TT_ + s] >= 0);
        const size_t idx = (size_t)m * N + n;
        const float v = acc[i][j][r] + bn + X[idx];  // residual
        X[idx] = keep ? v : 0.f;
      }
    }
  }
}

// ---------------------------------------------------------------------------
// Kernel 7: per-batch compaction of valid frame indices (stable order)
// ---------------------------------------------------------------------------
__global__ void build_order_kernel(const int* __restrict__ text,
                                   const unsigned char* __restrict__ amask,
                                   int* __restrict__ order,
                                   int* __restrict__ meta) {
  const int b = blockIdx.x;
  if (threadIdx.x != 0) return;
  int A = 0, cnt = 0;
  for (int s = 0; s < S_; ++s) {
    const bool am = amask[b * S_ + s] != 0;
    if (am) ++A;
    const bool pad = !((s < TT_) && (text[b * TT_ + s] >= 0));
    if (am && !pad) order[b * S_ + (cnt++)] = s;
  }
  meta[b * 2 + 0] = A;
  meta[b * 2 + 1] = cnt;
}

// ---------------------------------------------------------------------------
// Kernel 8: resampling gather -> out (B,S,D)
// ---------------------------------------------------------------------------
__global__ __launch_bounds__(256) void gather_kernel(
    const float* __restrict__ X, const int* __restrict__ order,
    const int* __restrict__ meta, float* __restrict__ out) {
  const int bt = blockIdx.x;
  const int b = bt >> 11, t = bt & (S_ - 1);
  const int A = meta[b * 2 + 0];
  const int Vn = meta[b * 2 + 1];
  const size_t obase = (size_t)bt * D_;
  const int t0 = threadIdx.x, t1 = threadIdx.x + 256;
  if (!((t < A) && (Vn > 0))) {
    out[obase + t0] = 0.f;
    out[obase + t1] = 0.f;
    return;
  }
  const int Vs = Vn;
  const int base = A / Vs, rem = A % Vs;
  const int nb = Vs - rem;
  const int split = nb * base;
  const int bm = base > 0 ? base : 1;
  int j = (t < split) ? (t / bm) : (nb + (t - split) / (base + 1));
  j = j < 0 ? 0 : (j > Vs - 1 ? Vs - 1 : j);
  const int src = order[b * S_ + j];
  const size_t ibase = ((size_t)b * S_ + src) * D_;
  out[obase + t0] = X[ibase + t0];
  out[obase + t1] = X[ibase + t1];
}

// ---------------------------------------------------------------------------
// Host driver. Workspace (all 16B aligned):
//   xbuf : M*D fp32 (64MB) | hbuf : M*I fp32 (128MB) | gx : B*I fp32
//   order: B*S int | meta : 2B int | w1t : L*I_*D_ bf16 | w2t : L*D_*I_ bf16
// Hnb (conv+LN bf16 output, M*D*2 = 32MB) lives in d_out (64MB), which is
// fully overwritten by gather_kernel at the end.
// ---------------------------------------------------------------------------
extern "C" void kernel_launch(void* const* d_in, const int* in_sizes, int n_in,
                              void* d_out, int out_size, void* d_ws, size_t ws_size,
                              hipStream_t stream) {
  const int* text            = (const int*)d_in[0];
  const unsigned char* amask = (const unsigned char*)d_in[1];
  // d_in[2] = seq_len scalar (== S_, fixed)
  const float* embed = (const float*)d_in[3];
  const float* freq  = (const float*)d_in[4];
  const float* dw_w  = (const float*)d_in[5];
  const float* dw_b  = (const float*)d_in[6];
  const float* ln_g  = (const float*)d_in[7];
  const float* ln_b  = (const float*)d_in[8];
  const float* w1    = (const float*)d_in[9];
  const float* b1    = (const float*)d_in[10];
  const float* grn_g = (const float*)d_in[11];
  const float* grn_b = (const float*)d_in[12];
  const float* w2    = (const float*)d_in[13];
  const float* b2    = (const float*)d_in[14];

  const size_t XN = (size_t)M_ * D_;
  const size_t HN = (size_t)M_ * I_;
  float* xbuf = (float*)d_ws;
  float* hbuf = xbuf + XN;
  float* gx   = hbuf + HN;
  int* order  = (int*)(gx + (size_t)B_ * I_);
  int* meta   = order + (size_t)B_ * S_;
  __bf16* w1t = (__bf16*)(meta + 2 * B_);            // (L, I_, D_) = N-major
  __bf16* w2t = w1t + (size_t)L_ * I_ * D_;          // (L, D_, I_)
  __bf16* hnb = (__bf16*)d_out;                      // M*D bf16 scratch

  embed_kernel<<<B_ * S_, 256, 0, stream>>>(text, embed, freq, xbuf);
  transpose_cvt_kernel<<<L_ * I_, 256, 0, stream>>>(w1, w1t, D_, I_);
  transpose_cvt_kernel<<<L_ * D_, 256, 0, stream>>>(w2, w2t, I_, D_);

  for (int l = 0; l < L_; ++l) {
    conv_ln_kernel<<<B_ * S_, 256, 0, stream>>>(
        xbuf, dw_w + (size_t)l * D_ * 7, dw_b + (size_t)l * D_,
        ln_g + (size_t)l * D_, ln_b + (size_t)l * D_, hnb);

    hipMemsetAsync(gx, 0, (size_t)B_ * I_ * sizeof(float), stream);

    gemm1_kernel<<<dim3(M_ / BM, I_ / BN), 256, 0, stream>>>(
        hnb, w1t + (size_t)l * I_ * D_, b1 + (size_t)l * I_, hbuf, gx);

    grn_nx_kernel<<<B_, 256, 0, stream>>>(gx);

    gemm2_kernel<<<dim3(M_ / BM, D_ / BN), 256, 0, stream>>>(
        hbuf, w2t + (size_t)l * D_ * I_, b2 + (size_t)l * D_,
        grn_g + (size_t)l * I_, grn_b + (size_t)l * I_, gx, text, xbuf);
  }

  build_order_kernel<<<B_, 32, 0, stream>>>(text, amask, order, meta);
  gather_kernel<<<B_ * S_, 256, 0, stream>>>(xbuf, order, meta, (float*)d_out);
}
